// ConvolutionCalculator_58841051955295
// MI455X (gfx1250) — matrix operand order repaired and checked by reading, not compiled, not run on hardware
//
#include <hip/hip_runtime.h>

typedef __attribute__((ext_vector_type(16))) _Float16 v16h;
typedef __attribute__((ext_vector_type(8)))  float    v8f;

#define C_XY   1.0925484305920792f   // 0.5  * sqrt(15/pi)
#define C_Z2   0.31539156525252005f  // 0.25 * sqrt(5/pi)
#define C_X2Y2 0.5462742152960396f   // 0.25 * sqrt(15/pi)

// One wave handles a tile of 16 edges:
//   A (16x32 f16)  = relu(dist * W1 + b1)   built directly in WMMA A layout
//   B (32x16 f16)  = W2                     loaded once, WMMA B layout
//   D (16x16 f32)  = A x B                  -> radial (+ b2)
//   out[e][c][q]   = radial[e][c] * sh[e][q]
__global__ __launch_bounds__(256) void edge_filter_wmma(
    const float* __restrict__ pos, const float* __restrict__ W1,
    const float* __restrict__ b1,  const float* __restrict__ W2,
    const float* __restrict__ b2,  const int* __restrict__ ei,
    float* __restrict__ out, long long E, long long tiles)
{
    __shared__ float sh_lds[8][80];              // per-wave 16 edges x 5 sh
    const int lane = threadIdx.x & 31;
    const int wib  = threadIdx.x >> 5;
    const long long wave = (long long)blockIdx.x * 8 + wib;
    const long long nw   = (long long)gridDim.x * 8;

    const int n     = lane & 15;   // output channel this lane owns in B/C/D
    const int hi    = lane >> 4;   // half-wave id
    const int base  = hi << 3;     // D row offset: 0 or 8
    const int khalf = hi << 4;     // B K half: 0 or 16

    // Per-lane slices of W1/b1 matching the A-matrix K ordering:
    // element idx of the v16h A register maps to k = khalf + idx
    float w1a[16], b1a[16];
    #pragma unroll
    for (int idx = 0; idx < 16; ++idx) {
        const int k = khalf + idx;
        w1a[idx] = W1[k];
        b1a[idx] = b1[k];
    }
    // B matrix (W2, 32x16), same K order as A: lane half owns K=khalf..khalf+15.
    v16h bmat;
    #pragma unroll
    for (int idx = 0; idx < 16; ++idx)
        bmat[idx] = (_Float16)W2[(khalf + idx) * 16 + n];
    const float b2v = b2[n];

    for (long long t = wave; t < tiles; t += nw) {
        const long long e0 = t << 4;
        const long long eL = e0 + (long long)(lane & 15);

        // Gather endpoints (positions table ~600 KB -> L2 resident)
        const int i = ei[eL];
        const int j = ei[E + eL];
        const float ax = pos[3*i], ay = pos[3*i+1], az = pos[3*i+2];
        const float bx = pos[3*j], by = pos[3*j+1], bz = pos[3*j+2];
        const float rx = ax - bx, ry = ay - by, rz = az - bz;
        float dist = sqrtf(rx*rx + ry*ry + rz*rz);
        dist = fmaxf(dist, 1e-8f);
        const float inv = 1.0f / dist;
        const float ux = rx*inv, uy = ry*inv, uz = rz*inv;

        // l=2 real spherical harmonics -> LDS (row m written by lane m)
        if (lane < 16) {
            float* s = &sh_lds[wib][lane * 5];
            s[0] = C_XY   * ux * uy;
            s[1] = C_XY   * uy * uz;
            s[2] = C_Z2   * (3.0f * uz * uz - 1.0f);
            s[3] = C_XY   * ux * uz;
            s[4] = C_X2Y2 * (ux * ux - uy * uy);
        }

        // Hidden layer packed straight into the WMMA A layout.
        v16h a;
        #pragma unroll
        for (int idx = 0; idx < 16; ++idx)
            a[idx] = (_Float16)fmaxf(fmaf(dist, w1a[idx], b1a[idx]), 0.0f);

        // D(16x16 f32) = A(16x32) x B(32x16): whole tile matmul in one op.
        v8f c = {};
        c = __builtin_amdgcn_wmma_f32_16x16x32_f16(
                /*neg_a=*/false, a, /*neg_b=*/false, bmat,
                /*c_mod=*/(short)0, c, /*reuse_a=*/false, /*reuse_b=*/false);

        __builtin_amdgcn_wave_barrier();  // keep LDS writes above the reads

        // VGPR r of D: lanes 0-15 -> M=r, lanes 16-31 -> M=8+r.
        #pragma unroll
        for (int r = 0; r < 8; ++r) {
            const int m = base + r;
            const float val = c[r] + b2v;
            const float* s = &sh_lds[wib][m * 5];
            float* o = out + ((size_t)(e0 + m) * 16 + (size_t)n) * 5;
            #pragma unroll
            for (int q = 0; q < 5; ++q)
                __builtin_nontemporal_store(val * s[q], o + q);
        }
    }
}

// Scalar fallback for E % 16 edges (not hit for E = 1.6M, kept for safety).
__global__ void edge_filter_tail(
    const float* __restrict__ pos, const float* __restrict__ W1,
    const float* __restrict__ b1,  const float* __restrict__ W2,
    const float* __restrict__ b2,  const int* __restrict__ ei,
    float* __restrict__ out, long long E, long long start)
{
    const long long e = start + (long long)blockIdx.x * blockDim.x + threadIdx.x;
    if (e >= E) return;
    const int i = ei[e];
    const int j = ei[E + e];
    const float rx = pos[3*i]   - pos[3*j];
    const float ry = pos[3*i+1] - pos[3*j+1];
    const float rz = pos[3*i+2] - pos[3*j+2];
    float dist = sqrtf(rx*rx + ry*ry + rz*rz);
    dist = fmaxf(dist, 1e-8f);
    const float inv = 1.0f / dist;
    const float ux = rx*inv, uy = ry*inv, uz = rz*inv;
    float sh[5];
    sh[0] = C_XY   * ux * uy;
    sh[1] = C_XY   * uy * uz;
    sh[2] = C_Z2   * (3.0f * uz * uz - 1.0f);
    sh[3] = C_XY   * ux * uz;
    sh[4] = C_X2Y2 * (ux * ux - uy * uy);
    float h[32];
    #pragma unroll
    for (int k = 0; k < 32; ++k)
        h[k] = fmaxf(fmaf(dist, W1[k], b1[k]), 0.0f);
    for (int c16 = 0; c16 < 16; ++c16) {
        float r = b2[c16];
        #pragma unroll
        for (int k = 0; k < 32; ++k)
            r = fmaf(h[k], W2[k * 16 + c16], r);
        float* o = out + ((size_t)e * 16 + c16) * 5;
        #pragma unroll
        for (int q = 0; q < 5; ++q)
            o[q] = r * sh[q];
    }
}

extern "C" void kernel_launch(void* const* d_in, const int* in_sizes, int n_in,
                              void* d_out, int out_size, void* d_ws, size_t ws_size,
                              hipStream_t stream) {
    (void)n_in; (void)out_size; (void)d_ws; (void)ws_size;
    const float* pos = (const float*)d_in[0];
    const float* W1  = (const float*)d_in[1];
    const float* b1  = (const float*)d_in[2];
    const float* W2  = (const float*)d_in[3];
    const float* b2  = (const float*)d_in[4];
    const int*   ei  = (const int*)d_in[5];
    float* out = (float*)d_out;

    const long long E     = (long long)in_sizes[5] / 2;   // edge_index is [2, E]
    const long long tiles = E / 16;
    const long long rem   = E - tiles * 16;

    if (tiles > 0) {
        long long blocks = (tiles + 7) / 8;               // 8 waves per 256-thr block
        if (blocks > 1048576LL) blocks = 1048576LL;       // grid-stride covers the rest
        edge_filter_wmma<<<(unsigned)blocks, 256, 0, stream>>>(
            pos, W1, b1, W2, b2, ei, out, E, tiles);
    }
    if (rem > 0) {
        edge_filter_tail<<<1, 32, 0, stream>>>(
            pos, W1, b1, W2, b2, ei, out, E, tiles * 16);
    }
}
